// Generator_20246475834098
// MI455X (gfx1250) — matrix operand order repaired and checked by reading, not compiled, not run on hardware
//
#include <hip/hip_runtime.h>
#include <hip/hip_bf16.h>

typedef __attribute__((ext_vector_type(16))) _Float16 v16h;
typedef __attribute__((ext_vector_type(8)))  _Float16 v8h;
typedef __attribute__((ext_vector_type(8)))  float    v8f;

namespace {
constexpr int B_   = 256;
constexpr int T_   = 128;
constexpr int H_   = 512;
constexpr int J_   = 22;
constexpr int DIN  = 3 * J_ + 4;              // 70
constexpr int DIN_PAD = 96;                   // padded K for x_t
constexpr int DDEC = 3 * J_ - 3 + DIN + H_;   // 645
constexpr int DDEC_PAD = 672;                 // padded K for ptc
constexpr int G3H  = 3 * H_;                  // 1536
constexpr int FC_OUT = 4 * (J_ + 1);          // 92
constexpr int FC_PAD = 96;
constexpr int PTC_PREV_OFF = 3 * J_ - 3;      // 63
constexpr int PTC_H_OFF    = PTC_PREV_OFF + DIN; // 133
}

// ---------------------------------------------------------------------------
// Pack W (N x K, row-major f32) into WMMA B-matrix fragments (f16).
// Fragment for (n-tile nt, k-step ks) is 512 halfs: lane-major, 16 halfs per
// lane.  Element (vgpr i, pair p) of lane l is
//   W[nt*16 + (l&15)][ks*32 + (l>>4)*16 + 2i + p]
// ---------------------------------------------------------------------------
__global__ void __launch_bounds__(32)
pack_weight_b_kernel(const float* __restrict__ W, _Float16* __restrict__ Wp,
                     int N, int K, int Kpad)
{
    const int l  = threadIdx.x;
    const int nt = blockIdx.x;
    const int kb = blockIdx.y;
    const int nK = Kpad >> 5;
    const int col = nt * 16 + (l & 15);
    _Float16* dst = Wp + ((size_t)nt * nK + kb) * 512 + l * 16;
#pragma unroll
    for (int i = 0; i < 8; ++i) {
        int k = kb * 32 + (l >> 4) * 16 + 2 * i;
        float v0 = (col < N && k     < K) ? W[(size_t)col * K + k]     : 0.0f;
        float v1 = (col < N && k + 1 < K) ? W[(size_t)col * K + k + 1] : 0.0f;
        dst[2 * i]     = (_Float16)v0;
        dst[2 * i + 1] = (_Float16)v1;
    }
}

// ---------------------------------------------------------------------------
// seqA_ (B,T,DIN) f32  ->  X16 [t][b][DIN_PAD] f16, zero padded
// ---------------------------------------------------------------------------
__global__ void __launch_bounds__(256)
convert_seq_kernel(const float* __restrict__ seq, _Float16* __restrict__ X16)
{
    int idx = blockIdx.x * 256 + threadIdx.x;
    const int total = T_ * B_ * DIN_PAD;
    if (idx >= total) return;
    int k  = idx % DIN_PAD;
    int bt = idx / DIN_PAD;
    int b  = bt % B_;
    int t  = bt / B_;
    float v = (k < DIN) ? seq[((size_t)b * T_ + t) * DIN + k] : 0.0f;
    X16[idx] = (_Float16)v;
}

__global__ void __launch_bounds__(256)
init_states_kernel(float* hA0, float* hA1, float* hB0, float* hB1,
                   _Float16* hA0h, _Float16* hA1h, _Float16* hB0h, _Float16* hB1h)
{
    int idx = blockIdx.x * 256 + threadIdx.x;   // over B_*H_
    hA0[idx] = 0.f; hA1[idx] = 0.f; hB0[idx] = 0.f; hB1[idx] = 0.f;
    hA0h[idx] = (_Float16)0.f; hA1h[idx] = (_Float16)0.f;
    hB0h[idx] = (_Float16)0.f; hB1h[idx] = (_Float16)0.f;
}

// ptc16 row = [skel_cond(63) | prev(70)=0 | hA1(512)=0 | pad(27)=0]
__global__ void __launch_bounds__(256)
init_ptc_kernel(const float* __restrict__ skelB, _Float16* __restrict__ ptc16)
{
    int idx = blockIdx.x * 256 + threadIdx.x;   // over B_*DDEC_PAD
    const int total = B_ * DDEC_PAD;
    if (idx >= total) return;
    int c = idx % DDEC_PAD;
    int b = idx / DDEC_PAD;
    float v = (c < PTC_PREV_OFF) ? skelB[(size_t)b * (3 * J_) + 3 + c] : 0.0f;
    ptc16[idx] = (_Float16)v;
}

// skel_in[b][j][k] = skelB[b][j*3+k] * dstd[j][k] + dmean[j][k]
__global__ void __launch_bounds__(256)
init_skelin_kernel(const float* __restrict__ skelB, const float* __restrict__ dstd,
                   const float* __restrict__ dmean, float* __restrict__ skel_in)
{
    int idx = blockIdx.x * 256 + threadIdx.x;   // over B_*66
    const int total = B_ * 3 * J_;
    if (idx >= total) return;
    int c = idx % (3 * J_);
    skel_in[idx] = skelB[idx] * dstd[c] + dmean[c];
}

// ---------------------------------------------------------------------------
// Dual GEMM in one launch: blockIdx.z==0 computes Ya = Xa*Wpa, z==1 computes
// Yb = Xb*Wpb.  One wave -> one 16x16 tile, K looped in steps of 32 via
// v_wmma_f32_16x16x32_f16.  Y is B_ x Npad f32, X is rows x Kpad f16.
// ---------------------------------------------------------------------------
__global__ void __launch_bounds__(128)
gemm_pair_kernel(const _Float16* __restrict__ Xa, const _Float16* __restrict__ Wpa,
                 float* __restrict__ Ya, int Kpa,
                 const _Float16* __restrict__ Xb, const _Float16* __restrict__ Wpb,
                 float* __restrict__ Yb, int Kpb, int Npad)
{
    const _Float16* X;
    const _Float16* Wp;
    float* Y;
    int Kpad;
    if (blockIdx.z == 0) { X = Xa; Wp = Wpa; Y = Ya; Kpad = Kpa; }
    else                 { X = Xb; Wp = Wpb; Y = Yb; Kpad = Kpb; }

    const int lane = threadIdx.x & 31;
    const int wave = threadIdx.x >> 5;
    const int nt = blockIdx.x * 4 + wave;     // 16-col tile index
    const int mt = blockIdx.y;                // 16-row tile index
    if (nt * 16 >= Npad) return;              // wave-uniform
    const int lm = lane & 15;
    const int hi = lane >> 4;
    const int nK = Kpad >> 5;

    const _Float16* Arow  = X  + ((size_t)(mt * 16 + lm)) * Kpad + hi * 16;
    const _Float16* Bfrag = Wp + ((size_t)nt * nK) * 512 + lane * 16;

    v8f acc = {0.f, 0.f, 0.f, 0.f, 0.f, 0.f, 0.f, 0.f};
    for (int ks = 0; ks < nK; ++ks) {
        union { v16h v; v8h h[2]; } a, b;
        a.h[0] = *(const v8h*)(Arow + ks * 32);       // K = hi*16 .. hi*16+7
        a.h[1] = *(const v8h*)(Arow + ks * 32 + 8);   // K = +8
        b.h[0] = *(const v8h*)(Bfrag + (size_t)ks * 512);
        b.h[1] = *(const v8h*)(Bfrag + (size_t)ks * 512 + 8);
        acc = __builtin_amdgcn_wmma_f32_16x16x32_f16(
            false, a.v, false, b.v, (short)0, acc, false, false);
    }
    // C layout: VGPR i -> row = hi*8 + i, col = lm
    float* Yr = Y + ((size_t)(mt * 16 + hi * 8)) * Npad + nt * 16 + lm;
#pragma unroll
    for (int i = 0; i < 8; ++i)
        Yr[(size_t)i * Npad] = acc[i];
}

// ---------------------------------------------------------------------------
// GRU gate combine: h' = (1-z)*n + z*h   (adds biases; writes f32 + f16 copies)
// Optionally also writes h' (f16) into an extra strided buffer (ptc slice).
// ---------------------------------------------------------------------------
__global__ void __launch_bounds__(256)
gru_combine_kernel(const float* __restrict__ gi, const float* __restrict__ gh,
                   const float* __restrict__ bih, const float* __restrict__ bhh,
                   float* __restrict__ h, _Float16* __restrict__ h16,
                   _Float16* __restrict__ extra16, int extra_stride, int extra_off)
{
    int idx = blockIdx.x * 256 + threadIdx.x;   // over B_*H_
    int b = idx >> 9;
    int j = idx & (H_ - 1);
    size_t r = (size_t)b * G3H;
    float ir  = gi[r + j]           + bih[j];
    float iz  = gi[r + H_ + j]      + bih[H_ + j];
    float in_ = gi[r + 2 * H_ + j]  + bih[2 * H_ + j];
    float hr  = gh[r + j]           + bhh[j];
    float hz  = gh[r + H_ + j]      + bhh[H_ + j];
    float hn  = gh[r + 2 * H_ + j]  + bhh[2 * H_ + j];
    float rg = 1.0f / (1.0f + expf(-(ir + hr)));
    float z  = 1.0f / (1.0f + expf(-(iz + hz)));
    float n  = tanhf(in_ + rg * hn);
    float hv = (1.0f - z) * n + z * h[idx];
    h[idx]   = hv;
    h16[idx] = (_Float16)hv;
    if (extra16)
        extra16[(size_t)b * extra_stride + extra_off + j] = (_Float16)hv;
}

// ---------------------------------------------------------------------------
// Fused FC head (WMMA) + forward kinematics + output writes + prev update.
// One block per 16 batch rows: 4 waves compute the 6 FC column tiles
// (wave-uniform tile assignment keeps EXEC all-ones for WMMA), results staged
// in LDS, then 16 threads walk the FK chain from LDS scratch.
// ---------------------------------------------------------------------------
__global__ void __launch_bounds__(128)
fc_fk_kernel(const _Float16* __restrict__ hX,     // B_ x H_ f16 (hB1h)
             const _Float16* __restrict__ WpFc,   // packed fc_W
             const float* __restrict__ fcb,       // FC_OUT
             const float* __restrict__ skel_in,   // B_ x J_ x 3
             const float* __restrict__ dmean, const float* __restrict__ dstd,
             const int* __restrict__ parents,
             float* __restrict__ out_loc,         // (pre-offset to step t) B_ x 66
             float* __restrict__ out_glob,        // B_ x 4
             float* __restrict__ out_quat,        // B_ x 88
             _Float16* __restrict__ ptc16)        // B_ x DDEC_PAD
{
    __shared__ float aos[16 * FC_PAD];            // FC result tile (16 x 96)
    __shared__ float gR[16 * J_ * 9];
    __shared__ float gP[16 * J_ * 3];

    const int mt   = blockIdx.x;                  // 16-row batch tile
    const int lane = threadIdx.x & 31;
    const int wave = threadIdx.x >> 5;
    const int lm = lane & 15;
    const int hi = lane >> 4;
    const int nK = H_ >> 5;                       // 16

    // wave w handles tiles {w, w+4} (w<2) or {w} -- uniform per wave
    for (int tt = 0; tt < 2; ++tt) {
        int nt = wave + tt * 4;
        if (nt >= FC_PAD / 16) break;
        const _Float16* Arow  = hX + ((size_t)(mt * 16 + lm)) * H_ + hi * 16;
        const _Float16* Bfrag = WpFc + ((size_t)nt * nK) * 512 + lane * 16;
        v8f acc = {0.f, 0.f, 0.f, 0.f, 0.f, 0.f, 0.f, 0.f};
        for (int ks = 0; ks < nK; ++ks) {
            union { v16h v; v8h h[2]; } a, b;
            a.h[0] = *(const v8h*)(Arow + ks * 32);
            a.h[1] = *(const v8h*)(Arow + ks * 32 + 8);
            b.h[0] = *(const v8h*)(Bfrag + (size_t)ks * 512);
            b.h[1] = *(const v8h*)(Bfrag + (size_t)ks * 512 + 8);
            acc = __builtin_amdgcn_wmma_f32_16x16x32_f16(
                false, a.v, false, b.v, (short)0, acc, false, false);
        }
#pragma unroll
        for (int i = 0; i < 8; ++i)
            aos[(hi * 8 + i) * FC_PAD + nt * 16 + lm] = acc[i];
    }
    __syncthreads();

    if (threadIdx.x < 16) {
        const int tid = threadIdx.x;
        const int b = mt * 16 + tid;
        const float* a = aos + tid * FC_PAD;
        float* myR = gR + tid * J_ * 9;
        float* myP = gP + tid * J_ * 3;
        float* bl = out_loc  + (size_t)b * (3 * J_);
        float* bq = out_quat + (size_t)b * (4 * J_);

        for (int j = 0; j < J_; ++j) {
            float qw = a[j * 4 + 0] + fcb[j * 4 + 0];
            float qx = a[j * 4 + 1] + fcb[j * 4 + 1];
            float qy = a[j * 4 + 2] + fcb[j * 4 + 2];
            float qz = a[j * 4 + 3] + fcb[j * 4 + 3];
            float inv = rsqrtf(qw * qw + qx * qx + qy * qy + qz * qz);
            float w = qw * inv, x = qx * inv, y = qy * inv, z = qz * inv;
            bq[j * 4 + 0] = w; bq[j * 4 + 1] = x;
            bq[j * 4 + 2] = y; bq[j * 4 + 3] = z;

            float x2 = x + x, y2 = y + y, z2 = z + z;
            float xx = x * x2, yy = y * y2, zz = z * z2;
            float xy = x * y2, yz = y * z2, xz = x * z2;
            float wx = w * x2, wy = w * y2, wz = w * z2;
            float R[9] = {1.0f - (yy + zz), xy - wz,          xz + wy,
                          xy + wz,          1.0f - (xx + zz), yz - wx,
                          xz - wy,          yz + wx,          1.0f - (xx + yy)};

            float s0 = skel_in[(size_t)b * (3 * J_) + j * 3 + 0];
            float s1 = skel_in[(size_t)b * (3 * J_) + j * 3 + 1];
            float s2 = skel_in[(size_t)b * (3 * J_) + j * 3 + 2];

            float pos[3];
            if (j == 0) {
#pragma unroll
                for (int e = 0; e < 9; ++e) myR[e] = R[e];
                pos[0] = s0; pos[1] = s1; pos[2] = s2;
            } else {
                int p = parents[j];
                const float* Rp = myR + p * 9;
#pragma unroll
                for (int r = 0; r < 3; ++r) {
#pragma unroll
                    for (int c = 0; c < 3; ++c)
                        myR[j * 9 + r * 3 + c] = Rp[r * 3 + 0] * R[0 + c] +
                                                 Rp[r * 3 + 1] * R[3 + c] +
                                                 Rp[r * 3 + 2] * R[6 + c];
                }
#pragma unroll
                for (int r = 0; r < 3; ++r)
                    pos[r] = Rp[r * 3 + 0] * s0 + Rp[r * 3 + 1] * s1 +
                             Rp[r * 3 + 2] * s2 + myP[p * 3 + r];
            }
            myP[j * 3 + 0] = pos[0]; myP[j * 3 + 1] = pos[1]; myP[j * 3 + 2] = pos[2];
#pragma unroll
            for (int k = 0; k < 3; ++k) {
                float v = (pos[k] - dmean[j * 3 + k]) / dstd[j * 3 + k];
                bl[j * 3 + k] = v;
                ptc16[(size_t)b * DDEC_PAD + PTC_PREV_OFF + j * 3 + k] = (_Float16)v;
            }
        }
#pragma unroll
        for (int k = 0; k < 4; ++k) {
            float g = a[4 * J_ + k] + fcb[4 * J_ + k];
            out_glob[(size_t)b * 4 + k] = g;
            ptc16[(size_t)b * DDEC_PAD + PTC_PREV_OFF + 3 * J_ + k] = (_Float16)g;
        }
    }
}

// ---------------------------------------------------------------------------
extern "C" void kernel_launch(void* const* d_in, const int* in_sizes, int n_in,
                              void* d_out, int out_size, void* d_ws, size_t ws_size,
                              hipStream_t stream)
{
    (void)in_sizes; (void)n_in; (void)out_size; (void)ws_size;
    const float* seqA   = (const float*)d_in[0];
    const float* skelB  = (const float*)d_in[2];
    const float* dmean  = (const float*)d_in[3];
    const float* dstd   = (const float*)d_in[4];
    const int*   parents= (const int*)d_in[6];
    const float* eWih0  = (const float*)d_in[7];
    const float* eWhh0  = (const float*)d_in[8];
    const float* ebih0  = (const float*)d_in[9];
    const float* ebhh0  = (const float*)d_in[10];
    const float* eWih1  = (const float*)d_in[11];
    const float* eWhh1  = (const float*)d_in[12];
    const float* ebih1  = (const float*)d_in[13];
    const float* ebhh1  = (const float*)d_in[14];
    const float* dWih0  = (const float*)d_in[15];
    const float* dWhh0  = (const float*)d_in[16];
    const float* dbih0  = (const float*)d_in[17];
    const float* dbhh0  = (const float*)d_in[18];
    const float* dWih1  = (const float*)d_in[19];
    const float* dWhh1  = (const float*)d_in[20];
    const float* dbih1  = (const float*)d_in[21];
    const float* dbhh1  = (const float*)d_in[22];
    const float* fcW    = (const float*)d_in[23];
    const float* fcb    = (const float*)d_in[24];

    char* ws = (char*)d_ws;
    auto alloc = [&](size_t bytes) -> void* {
        void* p = (void*)ws;
        ws += (bytes + 255) & ~(size_t)255;
        return p;
    };
    auto wp_halfs = [](int Kpad, int Npad) -> size_t {
        return (size_t)(Npad / 16) * (Kpad / 32) * 512;
    };

    _Float16* WpEih0 = (_Float16*)alloc(wp_halfs(DIN_PAD,  G3H)    * 2);
    _Float16* WpEhh0 = (_Float16*)alloc(wp_halfs(H_,       G3H)    * 2);
    _Float16* WpEih1 = (_Float16*)alloc(wp_halfs(H_,       G3H)    * 2);
    _Float16* WpEhh1 = (_Float16*)alloc(wp_halfs(H_,       G3H)    * 2);
    _Float16* WpDih0 = (_Float16*)alloc(wp_halfs(DDEC_PAD, G3H)    * 2);
    _Float16* WpDhh0 = (_Float16*)alloc(wp_halfs(H_,       G3H)    * 2);
    _Float16* WpDih1 = (_Float16*)alloc(wp_halfs(H_,       G3H)    * 2);
    _Float16* WpDhh1 = (_Float16*)alloc(wp_halfs(H_,       G3H)    * 2);
    _Float16* WpFc   = (_Float16*)alloc(wp_halfs(H_,       FC_PAD) * 2);
    _Float16* X16    = (_Float16*)alloc((size_t)T_ * B_ * DIN_PAD * 2);
    float* hA0 = (float*)alloc((size_t)B_ * H_ * 4);
    float* hA1 = (float*)alloc((size_t)B_ * H_ * 4);
    float* hB0 = (float*)alloc((size_t)B_ * H_ * 4);
    float* hB1 = (float*)alloc((size_t)B_ * H_ * 4);
    _Float16* hA0h = (_Float16*)alloc((size_t)B_ * H_ * 2);
    _Float16* hA1h = (_Float16*)alloc((size_t)B_ * H_ * 2);
    _Float16* hB0h = (_Float16*)alloc((size_t)B_ * H_ * 2);
    _Float16* hB1h = (_Float16*)alloc((size_t)B_ * H_ * 2);
    float* gi  = (float*)alloc((size_t)B_ * G3H * 4);
    float* gh  = (float*)alloc((size_t)B_ * G3H * 4);
    _Float16* ptc16 = (_Float16*)alloc((size_t)B_ * DDEC_PAD * 2);
    float* skel_in  = (float*)alloc((size_t)B_ * 3 * J_ * 4);

    // --- setup (all deterministic, re-run every call) ---
    auto pack = [&](const float* W, _Float16* Wp, int N, int K, int Kpad, int Npad) {
        dim3 g(Npad / 16, Kpad / 32);
        pack_weight_b_kernel<<<g, 32, 0, stream>>>(W, Wp, N, K, Kpad);
    };
    pack(eWih0, WpEih0, G3H,    DIN,  DIN_PAD,  G3H);
    pack(eWhh0, WpEhh0, G3H,    H_,   H_,       G3H);
    pack(eWih1, WpEih1, G3H,    H_,   H_,       G3H);
    pack(eWhh1, WpEhh1, G3H,    H_,   H_,       G3H);
    pack(dWih0, WpDih0, G3H,    DDEC, DDEC_PAD, G3H);
    pack(dWhh0, WpDhh0, G3H,    H_,   H_,       G3H);
    pack(dWih1, WpDih1, G3H,    H_,   H_,       G3H);
    pack(dWhh1, WpDhh1, G3H,    H_,   H_,       G3H);
    pack(fcW,   WpFc,   FC_OUT, H_,   H_,       FC_PAD);

    convert_seq_kernel<<<(T_ * B_ * DIN_PAD + 255) / 256, 256, 0, stream>>>(seqA, X16);
    init_states_kernel<<<(B_ * H_) / 256, 256, 0, stream>>>(hA0, hA1, hB0, hB1,
                                                            hA0h, hA1h, hB0h, hB1h);
    init_ptc_kernel<<<(B_ * DDEC_PAD + 255) / 256, 256, 0, stream>>>(skelB, ptc16);
    init_skelin_kernel<<<(B_ * 3 * J_ + 255) / 256, 256, 0, stream>>>(skelB, dstd, dmean, skel_in);

    auto gemm_pair = [&](const _Float16* Xa, const _Float16* Wpa, int Kpa,
                         const _Float16* Xb, const _Float16* Wpb, int Kpb) {
        dim3 g(G3H / 64, B_ / 16, 2);
        gemm_pair_kernel<<<g, 128, 0, stream>>>(Xa, Wpa, gi, Kpa,
                                                Xb, Wpb, gh, Kpb, G3H);
    };
    auto combine = [&](const float* bih, const float* bhh, float* h, _Float16* h16,
                       _Float16* extra, int estride, int eoff) {
        gru_combine_kernel<<<(B_ * H_) / 256, 256, 0, stream>>>(gi, gh, bih, bhh,
                                                                h, h16, extra, estride, eoff);
    };

    float* out_loc_base  = (float*)d_out;
    float* out_glob_base = out_loc_base  + (size_t)T_ * B_ * 3 * J_;
    float* out_quat_base = out_glob_base + (size_t)T_ * B_ * 4;

    for (int t = 0; t < T_; ++t) {
        // encoder layer 0 (gi & gh GEMMs fused into one launch)
        gemm_pair(X16 + (size_t)t * B_ * DIN_PAD, WpEih0, DIN_PAD,
                  hA0h, WpEhh0, H_);
        combine(ebih0, ebhh0, hA0, hA0h, nullptr, 0, 0);
        // encoder layer 1 (h also feeds decoder input slice of ptc)
        gemm_pair(hA0h, WpEih1, H_, hA1h, WpEhh1, H_);
        combine(ebih1, ebhh1, hA1, hA1h, ptc16, DDEC_PAD, PTC_H_OFF);
        // decoder layer 0
        gemm_pair(ptc16, WpDih0, DDEC_PAD, hB0h, WpDhh0, H_);
        combine(dbih0, dbhh0, hB0, hB0h, nullptr, 0, 0);
        // decoder layer 1
        gemm_pair(hB0h, WpDih1, H_, hB1h, WpDhh1, H_);
        combine(dbih1, dbhh1, hB1, hB1h, nullptr, 0, 0);
        // fused FC head + FK + outputs + prev for t+1
        fc_fk_kernel<<<B_ / 16, 128, 0, stream>>>(
            hB1h, WpFc, fcb, skel_in, dmean, dstd, parents,
            out_loc_base  + (size_t)t * B_ * 3 * J_,
            out_glob_base + (size_t)t * B_ * 4,
            out_quat_base + (size_t)t * B_ * 4 * J_,
            ptc16);
    }
}
